// AgentAttention_26276609917670
// MI455X (gfx1250) — matrix operand order repaired and checked by reading, not compiled, not run on hardware
//
#include <hip/hip_runtime.h>
#include <hip/hip_bf16.h>

typedef __attribute__((ext_vector_type(16))) _Float16 v16h;
typedef __attribute__((ext_vector_type(8)))  _Float16 v8h;
typedef __attribute__((ext_vector_type(8)))  float    v8f;

#define NH 16
#define DH 64
#define A_NUM 50
#define S_LEN 512
#define D_MODEL 1024
#define BS 16
#define SCALE_ATT 0.125f
#define M_ROWS (BS * S_LEN)   // 8192

union Frag { v16h v; v8h h[2]; };

// ---- CDNA5 async copy to LDS (global_load_async_to_lds_b128) with safe fallback ----
#if defined(__has_builtin)
#if __has_builtin(__builtin_amdgcn_global_load_async_to_lds_b128)
#define HAVE_ASYNC_COPY 1
#endif
#endif

#ifdef HAVE_ASYNC_COPY
typedef int v4i_ __attribute__((vector_size(16)));
typedef __attribute__((address_space(1))) v4i_ glb_v4i;
typedef __attribute__((address_space(3))) v4i_ lds_v4i;
__device__ __forceinline__ void async_cp16(void* l, const void* g) {
    __builtin_amdgcn_global_load_async_to_lds_b128((glb_v4i*)g, (lds_v4i*)l, 0, 0);
}
__device__ __forceinline__ void async_wait0() {
#if __has_builtin(__builtin_amdgcn_s_wait_asynccnt)
    __builtin_amdgcn_s_wait_asynccnt(0);
#else
    asm volatile("s_wait_asynccnt 0x0" ::: "memory");
#endif
}
#else
__device__ __forceinline__ void async_cp16(void* l, const void* g) {
    *(uint4*)l = *(const uint4*)g;
}
__device__ __forceinline__ void async_wait0() {}
#endif

// ---- WMMA fragment loaders (CDNA5 16x16x32 f16 layouts) ----
// A (16x32, MxK): lane = m + 16*khalf; elems 0..7 -> K = 8*kh + j ; elems 8..15 -> K = 16 + 8*kh + (j-8)
__device__ __forceinline__ v16h load_frag_a(const _Float16* base, int row0, int k0, int ld) {
    int lane = threadIdx.x & 31;
    int m = lane & 15, kh = lane >> 4;
    const _Float16* p = base + (size_t)(row0 + m) * ld + k0 + 8 * kh;
    Frag f;
    f.h[0] = *(const v8h*)(p);
    f.h[1] = *(const v8h*)(p + 16);
    return f.v;
}
// B (32x16, KxN) loaded from transposed storage Bt[n][k]: lane = n + 16*khalf; elems 0..7 -> K = 8*kh + j ; elems 8..15 -> K = 16 + 8*kh + (j-8)
__device__ __forceinline__ v16h load_frag_b(const _Float16* base, int n0, int k0, int ld) {
    int lane = threadIdx.x & 31;
    int n = lane & 15, kh = lane >> 4;
    const _Float16* p = base + (size_t)(n0 + n) * ld + k0 + 8 * kh;
    Frag f;
    f.h[0] = *(const v8h*)(p);
    f.h[1] = *(const v8h*)(p + 16);
    return f.v;
}

__device__ __forceinline__ v8f wmma_f16(v16h a, v16h b, v8f c) {
    return __builtin_amdgcn_wmma_f32_16x16x32_f16(false, a, false, b, (short)0, c, false, false);
}

// ============================ conversion kernels ============================
__global__ __launch_bounds__(256) void cvt_x_f16(const float* __restrict__ x,
                                                 _Float16* __restrict__ xh, int n) {
    int i = blockIdx.x * 256 + threadIdx.x;
    if (i < n) xh[i] = (_Float16)x[i];
}

__global__ __launch_bounds__(256) void cvt_w_t_f16(const float* __restrict__ Wq,
                                                   const float* __restrict__ Wk,
                                                   const float* __restrict__ Wv,
                                                   _Float16* __restrict__ WtBase) {
    int z = blockIdx.y;
    const float* W = (z == 0) ? Wq : (z == 1) ? Wk : Wv;
    _Float16* Wt = WtBase + (size_t)z * D_MODEL * D_MODEL;
    int i = blockIdx.x * 256 + threadIdx.x;   // i = k*1024 + n (coalesced read)
    if (i < D_MODEL * D_MODEL) {
        int k = i >> 10, n = i & 1023;
        Wt[(size_t)n * D_MODEL + k] = (_Float16)W[i];
    }
}

// ============================ QKV projection GEMM ============================
// C[8192,1024] = Xh[8192,1024] @ W[1024,1024] + bias (W transposed: Wt[n][k])
// Writes f32 output AND an f16 mirror for the attention-stage WMMAs.
__global__ __launch_bounds__(256) void proj_gemm(const _Float16* __restrict__ X,
                                                 const _Float16* __restrict__ WtBase,
                                                 const float* __restrict__ bq,
                                                 const float* __restrict__ bk,
                                                 const float* __restrict__ bv,
                                                 float* __restrict__ outBase,
                                                 _Float16* __restrict__ out16Base) {
    int z = blockIdx.z;
    const _Float16* Wt = WtBase + (size_t)z * D_MODEL * D_MODEL;
    const float* bias = (z == 0) ? bq : (z == 1) ? bk : bv;
    float* out = outBase + (size_t)z * M_ROWS * D_MODEL;
    _Float16* out16 = out16Base + (size_t)z * M_ROWS * D_MODEL;

    int w = threadIdx.x >> 5;
    int wm = w >> 1, wn = w & 1;
    int row0 = blockIdx.x * 128 + wm * 32;
    int col0 = blockIdx.y * 128 + wn * 64;

    v8f acc[2][4] = {};
    for (int k0 = 0; k0 < D_MODEL; k0 += 32) {
        v16h aF[2], bF[4];
#pragma unroll
        for (int mi = 0; mi < 2; mi++) aF[mi] = load_frag_a(X, row0 + 16 * mi, k0, D_MODEL);
#pragma unroll
        for (int ni = 0; ni < 4; ni++) bF[ni] = load_frag_b(Wt, col0 + 16 * ni, k0, D_MODEL);
        if (k0 + 32 < D_MODEL) {   // L2 prefetch of next K tile -> global_prefetch_b8
            __builtin_prefetch(X + (size_t)(row0 + (threadIdx.x & 31)) * D_MODEL + k0 + 32, 0, 1);
            __builtin_prefetch(Wt + (size_t)(col0 + (threadIdx.x & 31)) * D_MODEL + k0 + 32, 0, 1);
        }
#pragma unroll
        for (int mi = 0; mi < 2; mi++)
#pragma unroll
            for (int ni = 0; ni < 4; ni++) acc[mi][ni] = wmma_f16(aF[mi], bF[ni], acc[mi][ni]);
    }
    int lane = threadIdx.x & 31;
    int cn = lane & 15, kh = lane >> 4;
#pragma unroll
    for (int mi = 0; mi < 2; mi++)
#pragma unroll
        for (int ni = 0; ni < 4; ni++) {
            int col = col0 + 16 * ni + cn;
            float bvv = bias[col];
#pragma unroll
            for (int i = 0; i < 8; i++) {
                int row = row0 + 16 * mi + i + 8 * kh;
                float val = acc[mi][ni][i] + bvv;
                out[(size_t)row * D_MODEL + col] = val;
                out16[(size_t)row * D_MODEL + col] = (_Float16)val;
            }
        }
}

// ============================ agent interpolation ============================
__global__ __launch_bounds__(256) void interp_agents(const float* __restrict__ Q,
                                                     float* __restrict__ agents) {
    int idx = blockIdx.x * 256 + threadIdx.x;       // [b][a][c]
    if (idx >= BS * A_NUM * D_MODEL) return;
    int c = idx & 1023;
    int a = (idx >> 10) % A_NUM;
    int b = idx / (A_NUM * D_MODEL);
    float src = (a + 0.5f) * ((float)S_LEN / (float)A_NUM) - 0.5f;
    if (src < 0.f) src = 0.f;
    int i0 = (int)floorf(src);
    if (i0 > S_LEN - 1) i0 = S_LEN - 1;
    int i1 = (i0 + 1 < S_LEN) ? i0 + 1 : S_LEN - 1;
    float wg = src - (float)i0;
    agents[idx] = Q[((size_t)b * S_LEN + i0) * D_MODEL + c] * (1.f - wg)
                + Q[((size_t)b * S_LEN + i1) * D_MODEL + c] * wg;
}

// ============================ V transpose (f16): Vt16[b,h,d,s] ============================
__global__ __launch_bounds__(256) void build_vt(const _Float16* __restrict__ V16,
                                                _Float16* __restrict__ Vt16) {
    int idx = blockIdx.x * 256 + threadIdx.x;       // [bh][d][s]
    if (idx >= BS * NH * DH * S_LEN) return;
    int s = idx & 511;
    int d = (idx >> 9) & 63;
    int bh = idx >> 15;
    int b = bh >> 4, h = bh & 15;
    Vt16[idx] = V16[((size_t)b * S_LEN + s) * D_MODEL + h * DH + d];
}

// ============================ stage 1: agents attend keys ============================
// LDS: kS f16[512][72], vT f16[64][520], sS f32[64][512], aS f16[64][72]; probs reuse kS (ld 520)
#define S1_KS   0
#define S1_VT   73728
#define S1_SS   (73728 + 66560)
#define S1_AS   (73728 + 66560 + 131072)
#define S1_SHM  (73728 + 66560 + 131072 + 9216)

__global__ __launch_bounds__(256) void stage1(const float* __restrict__ agents,
                                              const _Float16* __restrict__ K16,
                                              const _Float16* __restrict__ Vt16,
                                              const float* __restrict__ emb,
                                              const float* __restrict__ mask,
                                              float* __restrict__ agent_v) {
    extern __shared__ __align__(16) char smem[];
    _Float16* kS = (_Float16*)(smem + S1_KS);   // [512][72]
    _Float16* vT = (_Float16*)(smem + S1_VT);   // [64][520]  vT[d][r]
    float*    sS = (float*)   (smem + S1_SS);   // [64][512]
    _Float16* aS = (_Float16*)(smem + S1_AS);   // [64][72]
    _Float16* pS = (_Float16*)(smem + S1_KS);   // probs [64][520] (reuses kS)

    int bh = blockIdx.x, b = bh >> 4, h = bh & 15;
    const _Float16* Kh = K16 + (size_t)b * S_LEN * D_MODEL + h * DH;
    const _Float16* Vth = Vt16 + (size_t)bh * DH * S_LEN;
    const float* Ah = agents + (size_t)b * A_NUM * D_MODEL + h * DH;
    const float* maskb = mask + (size_t)b * S_LEN;
    int w = threadIdx.x >> 5;
    int lane = threadIdx.x & 31;

    // async-stage K rows (128B each, 8x16B chunks) into padded LDS (ld=72 f16 -> 144B rows)
    for (int i = threadIdx.x; i < 4096; i += 256) {
        int r = i >> 3, ch = i & 7;
        async_cp16((char*)kS + r * 144 + ch * 16,
                   (const char*)Kh + (size_t)r * 2048 + ch * 16);
    }
    // async-stage V^T rows (1024B each, 64x16B chunks) into padded LDS (ld=520 f16 -> 1040B rows)
    for (int i = threadIdx.x; i < 4096; i += 256) {
        int d = i >> 6, ch = i & 63;
        async_cp16((char*)vT + d * 1040 + ch * 16,
                   (const char*)Vth + (size_t)d * 1024 + ch * 16);
    }
    // agents: small, convert f32 -> f16 with zero padding rows 50..63
    for (int i = threadIdx.x; i < 64 * DH; i += 256) {
        int a = i >> 6, d = i & 63;
        aS[a * 72 + d] = (_Float16)((a < A_NUM) ? Ah[(size_t)a * D_MODEL + d] : 0.0f);
    }
    async_wait0();
    __syncthreads();

    // scores = SCALE * agents @ K^T : 4 M-tiles x 32 N-tiles, K=64
    for (int t = w; t < 128; t += 8) {
        int mt = t >> 5, nt = t & 31;
        v8f acc = {};
#pragma unroll
        for (int kt = 0; kt < 64; kt += 32)
            acc = wmma_f16(load_frag_a(aS, mt * 16, kt, 72),
                           load_frag_b(kS, nt * 16, kt, 72), acc);
        int cn = lane & 15, kh = lane >> 4;
#pragma unroll
        for (int i = 0; i < 8; i++)
            sS[(mt * 16 + i + 8 * kh) * S_LEN + nt * 16 + cn] = acc[i] * SCALE_ATT;
    }
    __syncthreads();

    // relative-position bias: dot(agents[l]+k[r], emb[l-r+511]) + mask
    for (int i = threadIdx.x; i < A_NUM * S_LEN; i += 256) {
        int l = i >> 9, r = i & 511;
        const float* e = emb + (size_t)(l - r + 511) * DH;
        float dot = 0.f;
#pragma unroll 8
        for (int d = 0; d < DH; d++)
            dot += ((float)aS[l * 72 + d] + (float)kS[r * 72 + d]) * e[d];
        sS[l * S_LEN + r] += dot + maskb[r];
    }
    __syncthreads();

    // softmax over r (wave per row), write f16 probs into pS (overwrites kS)
    for (int l = w; l < A_NUM; l += 8) {
        float mx = -1e30f;
        for (int r = lane; r < S_LEN; r += 32) mx = fmaxf(mx, sS[l * S_LEN + r]);
        for (int o = 16; o; o >>= 1) mx = fmaxf(mx, __shfl_xor(mx, o, 32));
        float sum = 0.f;
        for (int r = lane; r < S_LEN; r += 32) sum += __expf(sS[l * S_LEN + r] - mx);
        for (int o = 16; o; o >>= 1) sum += __shfl_xor(sum, o, 32);
        float inv = 1.0f / sum;
        for (int r = lane; r < S_LEN; r += 32)
            pS[l * 520 + r] = (_Float16)(__expf(sS[l * S_LEN + r] - mx) * inv);
    }
    for (int i = threadIdx.x; i < (64 - A_NUM) * S_LEN; i += 256) {
        int l = A_NUM + i / S_LEN, r = i % S_LEN;
        pS[l * 520 + r] = (_Float16)0.f;
    }
    __syncthreads();

    // agent_v = probs @ V : 4x4 tiles, K=512
    for (int t = w; t < 16; t += 8) {
        int mt = t >> 2, nt = t & 3;
        v8f acc = {};
        for (int kt = 0; kt < S_LEN; kt += 32)
            acc = wmma_f16(load_frag_a(pS, mt * 16, kt, 520),
                           load_frag_b(vT, nt * 16, kt, 520), acc);
        int cn = lane & 15, kh = lane >> 4;
#pragma unroll
        for (int i = 0; i < 8; i++)
            agent_v[((size_t)bh * 64 + mt * 16 + i + 8 * kh) * DH + nt * 16 + cn] = acc[i];
    }
}

// ============================ stage 2: queries attend agents ============================
// LDS: qS f16[512][72], aS f16[64][72], avT f16[64][72], sS f32[512][64]; probs reuse qS
#define S2_QS   0
#define S2_AS   73728
#define S2_AVT  (73728 + 9216)
#define S2_SS   (73728 + 9216 + 9216)
#define S2_SHM  (73728 + 9216 + 9216 + 131072)

__global__ __launch_bounds__(256) void stage2(const _Float16* __restrict__ Q16,
                                              const float* __restrict__ agents,
                                              const float* __restrict__ agent_v,
                                              const float* __restrict__ emb,
                                              const float* __restrict__ mask,
                                              float* __restrict__ out) {
    extern __shared__ __align__(16) char smem[];
    _Float16* qS  = (_Float16*)(smem + S2_QS);   // [512][72]
    _Float16* aS  = (_Float16*)(smem + S2_AS);   // [64][72]
    _Float16* avT = (_Float16*)(smem + S2_AVT);  // [64][72]  avT[d][a]
    float*    sS  = (float*)   (smem + S2_SS);   // [512][64]
    _Float16* pS  = (_Float16*)(smem + S2_QS);   // probs [512][72] (reuses qS)

    int bh = blockIdx.x, b = bh >> 4, h = bh & 15;
    const _Float16* Qh = Q16 + (size_t)b * S_LEN * D_MODEL + h * DH;
    const float* Ah = agents + (size_t)b * A_NUM * D_MODEL + h * DH;
    const float* AVh = agent_v + (size_t)bh * 64 * DH;
    const float* maskb = mask + (size_t)b * S_LEN;
    int w = threadIdx.x >> 5;
    int lane = threadIdx.x & 31;

    // async-stage Q rows into padded LDS
    for (int i = threadIdx.x; i < 4096; i += 256) {
        int s = i >> 3, ch = i & 7;
        async_cp16((char*)qS + s * 144 + ch * 16,
                   (const char*)Qh + (size_t)s * 2048 + ch * 16);
    }
    for (int i = threadIdx.x; i < 64 * DH; i += 256) {
        int a = i >> 6, d = i & 63;
        aS[a * 72 + d]  = (_Float16)((a < A_NUM) ? Ah[(size_t)a * D_MODEL + d] : 0.0f);
        avT[d * 72 + a] = (_Float16)((a < A_NUM) ? AVh[(size_t)a * DH + d] : 0.0f);
    }
    async_wait0();
    __syncthreads();

    // scores = SCALE * Q @ agents^T : 32 M-tiles x 4 N-tiles, K=64
    for (int t = w; t < 128; t += 8) {
        int mt = t >> 2, nt = t & 3;
        v8f acc = {};
#pragma unroll
        for (int kt = 0; kt < 64; kt += 32)
            acc = wmma_f16(load_frag_a(qS, mt * 16, kt, 72),
                           load_frag_b(aS, nt * 16, kt, 72), acc);
        int cn = lane & 15, kh = lane >> 4;
#pragma unroll
        for (int i = 0; i < 8; i++)
            sS[(mt * 16 + i + 8 * kh) * 64 + nt * 16 + cn] = acc[i] * SCALE_ATT;
    }
    __syncthreads();

    // bias: dot(q[s]+agents[a], emb[s-a+511]) + mask^T (per-row constant)
    for (int i = threadIdx.x; i < S_LEN * A_NUM; i += 256) {
        int s = i / A_NUM, a = i % A_NUM;
        const float* e = emb + (size_t)(s - a + 511) * DH;
        float dot = 0.f;
#pragma unroll 8
        for (int d = 0; d < DH; d++)
            dot += ((float)qS[s * 72 + d] + (float)aS[a * 72 + d]) * e[d];
        sS[s * 64 + a] += dot + maskb[s];
    }
    __syncthreads();

    // per-row softmax over 50 agents; probs overwrite qS region
    for (int s = threadIdx.x; s < S_LEN; s += 256) {
        float mx = -1e30f;
        for (int a = 0; a < A_NUM; a++) mx = fmaxf(mx, sS[s * 64 + a]);
        float sum = 0.f;
        for (int a = 0; a < A_NUM; a++) sum += __expf(sS[s * 64 + a] - mx);
        float inv = 1.0f / sum;
        for (int a = 0; a < A_NUM; a++)
            pS[s * 72 + a] = (_Float16)(__expf(sS[s * 64 + a] - mx) * inv);
        for (int a = A_NUM; a < 64; a++) pS[s * 72 + a] = (_Float16)0.f;
    }
    __syncthreads();

    // x = probs @ agent_v : 32 M-tiles x 4 N-tiles, K=64
    for (int t = w; t < 128; t += 8) {
        int mt = t >> 2, nt = t & 3;
        v8f acc = {};
#pragma unroll
        for (int kt = 0; kt < 64; kt += 32)
            acc = wmma_f16(load_frag_a(pS, mt * 16, kt, 72),
                           load_frag_b(avT, nt * 16, kt, 72), acc);
        int cn = lane & 15, kh = lane >> 4;
#pragma unroll
        for (int i = 0; i < 8; i++) {
            int s = mt * 16 + i + 8 * kh;
            out[((size_t)b * S_LEN + s) * D_MODEL + h * DH + nt * 16 + cn] = acc[i];
        }
    }
}

// ============================ depthwise 3x3 conv epilogue ============================
__global__ __launch_bounds__(256) void conv_add(const float* __restrict__ V,
                                                const float* __restrict__ w9,
                                                const float* __restrict__ bconv,
                                                float* __restrict__ out) {
    size_t idx = (size_t)blockIdx.x * 256 + threadIdx.x;
    if (idx >= (size_t)BS * S_LEN * D_MODEL) return;
    int c = (int)(idx & 1023);
    int s = (int)((idx >> 10) & 511);
    int b = (int)(idx >> 19);
    float acc = bconv[0];
#pragma unroll
    for (int i = 0; i < 3; i++) {
        int ss = s - 1 + i;
        if (ss < 0 || ss >= S_LEN) continue;
#pragma unroll
        for (int j = 0; j < 3; j++) {
            int cc = c - 1 + j;
            if (cc < 0 || cc >= D_MODEL) continue;
            acc += w9[i * 3 + j] * V[((size_t)b * S_LEN + ss) * D_MODEL + cc];
        }
    }
    out[idx] += acc;
}

// ============================ host launcher ============================
extern "C" void kernel_launch(void* const* d_in, const int* in_sizes, int n_in,
                              void* d_out, int out_size, void* d_ws, size_t ws_size,
                              hipStream_t stream) {
    const float* hidden = (const float*)d_in[0];
    const float* mask   = (const float*)d_in[1];
    const float* Wq     = (const float*)d_in[2];
    const float* bq     = (const float*)d_in[3];
    const float* Wk     = (const float*)d_in[4];
    const float* bk     = (const float*)d_in[5];
    const float* Wv     = (const float*)d_in[6];
    const float* bv     = (const float*)d_in[7];
    const float* emb    = (const float*)d_in[8];
    const float* dwc_w  = (const float*)d_in[9];
    const float* dwc_b  = (const float*)d_in[10];
    float* out = (float*)d_out;

    char* ws = (char*)d_ws;
    const size_t offXh  = 0;                                          // f16 X        16 MB
    const size_t offWt  = offXh + (size_t)M_ROWS * D_MODEL * 2;       // f16 Wt x3     6 MB
    const size_t offQ   = offWt + (size_t)3 * D_MODEL * D_MODEL * 2;  // f32 Q,K,V    96 MB
    const size_t offAg  = offQ + (size_t)3 * M_ROWS * D_MODEL * 4;    // f32 agents
    const size_t offAv  = offAg + (size_t)BS * A_NUM * D_MODEL * 4;   // f32 agent_v
    const size_t offQ16 = offAv + (size_t)BS * NH * 64 * DH * 4;      // f16 Q,K,V    48 MB
    const size_t offVt  = offQ16 + (size_t)3 * M_ROWS * D_MODEL * 2;  // f16 V^T      16 MB

    _Float16* Xh = (_Float16*)(ws + offXh);
    _Float16* Wt = (_Float16*)(ws + offWt);
    float* Qb = (float*)(ws + offQ);
    float* Kb = Qb + (size_t)M_ROWS * D_MODEL;
    float* Vb = Kb + (size_t)M_ROWS * D_MODEL;
    float* Ag = (float*)(ws + offAg);
    float* Av = (float*)(ws + offAv);
    _Float16* Q16 = (_Float16*)(ws + offQ16);
    _Float16* K16 = Q16 + (size_t)M_ROWS * D_MODEL;
    _Float16* V16 = K16 + (size_t)M_ROWS * D_MODEL;
    _Float16* Vt16 = (_Float16*)(ws + offVt);

    cvt_x_f16<<<(M_ROWS * D_MODEL + 255) / 256, 256, 0, stream>>>(hidden, Xh, M_ROWS * D_MODEL);
    cvt_w_t_f16<<<dim3((D_MODEL * D_MODEL + 255) / 256, 3), 256, 0, stream>>>(Wq, Wk, Wv, Wt);
    proj_gemm<<<dim3(M_ROWS / 128, D_MODEL / 128, 3), 256, 0, stream>>>(Xh, Wt, bq, bk, bv, Qb, Q16);
    interp_agents<<<(BS * A_NUM * D_MODEL + 255) / 256, 256, 0, stream>>>(Qb, Ag);
    build_vt<<<(BS * NH * DH * S_LEN + 255) / 256, 256, 0, stream>>>(V16, Vt16);
    stage1<<<BS * NH, 256, S1_SHM, stream>>>(Ag, K16, Vt16, emb, mask, Av);
    stage2<<<BS * NH, 256, S2_SHM, stream>>>(Q16, Ag, Av, emb, mask, out);
    conv_add<<<(BS * S_LEN * D_MODEL + 255) / 256, 256, 0, stream>>>(Vb, dwc_w, dwc_b, out);
}
